// PatternGraphSAGE_17102559773409
// MI455X (gfx1250) — matrix operand order repaired and checked by reading, not compiled, not run on hardware
//
#include <hip/hip_runtime.h>
#include <hip/hip_bf16.h>

// ---------------------------------------------------------------------------
// GraphSAGE forward for MI455X (gfx1250, wave32).
// Sparse aggregation: f32 atomics into L2-resident agg buffers (h fits in
// the 192MB L2, so gathers are L2-served). Dense lin_l/lin_r: bf16 WMMA
// (v_wmma_f32_16x16x32_bf16), 4 M-tiles per wave with a shared, K-contiguous
// B fragment loaded from pre-transposed bf16 weights.
// ---------------------------------------------------------------------------

typedef __attribute__((ext_vector_type(16))) __bf16 v16bf;
typedef __attribute__((ext_vector_type(8)))  __bf16 v8bf;
typedef __attribute__((ext_vector_type(4)))  __bf16 v4bf;
typedef __attribute__((ext_vector_type(8)))  float  v8f;

#define N_NODES  50000
#define N_PAD    50048   // 64 * 782 : padded M so WMMA waves need no guards
#define N_EDGES  800000
#define N_GRAPHS 64

// ---------------- degree + layer-0 scalar aggregation ----------------------
__global__ void k_deg_agg0(const int* __restrict__ src, const int* __restrict__ dst,
                           const float* __restrict__ x,
                           float* __restrict__ deg, float* __restrict__ agg0) {
    int e = blockIdx.x * blockDim.x + threadIdx.x;
    if (e >= N_EDGES) return;
    int s = src[e], d = dst[e];
    atomicAdd(deg + d, 1.0f);
    atomicAdd(agg0 + d, x[s]);
}

__global__ void k_deg_inv(const float* __restrict__ deg, float* __restrict__ dinv) {
    int n = blockIdx.x * blockDim.x + threadIdx.x;
    if (n >= N_PAD) return;
    dinv[n] = 1.0f / fmaxf(deg[n], 1.0f);   // padded rows: deg=0 -> dinv=1
}

// layer 0: K=1, pure outer product -> scalar epilogue, output bf16 (relu)
__global__ void k_layer0(const float* __restrict__ x, const float* __restrict__ agg0,
                         const float* __restrict__ dinv, const float* __restrict__ Wl,
                         const float* __restrict__ b, const float* __restrict__ Wr,
                         __bf16* __restrict__ h0) {
    int idx = blockIdx.x * blockDim.x + threadIdx.x;
    if (idx >= N_PAD * 256) return;
    int n = idx >> 8, c = idx & 255;
    if (n >= N_NODES) { h0[idx] = (__bf16)0.0f; return; }
    float v = agg0[n] * dinv[n] * Wl[c] + b[c] + x[n] * Wr[c];
    h0[idx] = (__bf16)fmaxf(v, 0.0f);
}

// ---------------- weight f32[CIN][COUT] -> bf16 W^T[COUT][CIN] -------------
__global__ void k_w_transpose(const float* __restrict__ in, __bf16* __restrict__ out,
                              int cin, int cout) {
    int i = blockIdx.x * blockDim.x + threadIdx.x;
    if (i >= cin * cout) return;
    int k = i / cout, c = i - k * cout;
    out[(size_t)c * cin + k] = (__bf16)in[i];
}

// ---------------- feature scatter: agg[dst] += h[src] (C = 256) ------------
__global__ void k_scatter256(const int* __restrict__ src, const int* __restrict__ dst,
                             const __bf16* __restrict__ h, float* __restrict__ agg) {
    int idx = blockIdx.x * blockDim.x + threadIdx.x;   // N_EDGES * 64 threads
    if (idx >= N_EDGES * 64) return;
    int e = idx >> 6;
    int c = (idx & 63) << 2;
    int s = src[e], d = dst[e];
    v4bf hv = *(const v4bf*)(h + (size_t)s * 256 + c); // one b64 load
    float* ap = agg + (size_t)d * 256 + c;
    atomicAdd(ap + 0, (float)hv[0]);
    atomicAdd(ap + 1, (float)hv[1]);
    atomicAdd(ap + 2, (float)hv[2]);
    atomicAdd(ap + 3, (float)hv[3]);
}

// ---------------- WMMA fragment helpers ------------------------------------
// A-frag (16-bit A 16x32, ISA 7.12.2): lane<16 -> M=lane, K in
//   {kb..kb+7} u {16+kb..23+kb}, kb = (lane>=16)*8.  p pre-offset by kb.
__device__ inline v16bf load_a_bf(const __bf16* p) {
    v8bf lo = *(const v8bf*)(p);        // 16B
    v8bf hi = *(const v8bf*)(p + 16);   // 16B
    return __builtin_shufflevector(lo, hi, 0,1,2,3,4,5,6,7,8,9,10,11,12,13,14,15);
}
__device__ inline v16bf load_a_f32_scaled(const float* p, float s) {
    v8f lo = *(const v8f*)(p);          // 32B
    v8f hi = *(const v8f*)(p + 16);     // 32B
    v16bf a;
#pragma unroll
    for (int e = 0; e < 8; ++e) {
        a[e]     = (__bf16)(lo[e] * s);
        a[e + 8] = (__bf16)(hi[e] * s);
    }
    return a;
}

// ---------------- fused dual-GEMM SAGE layer (WMMA bf16) -------------------
// D = relu?( bf16(agg * dinv) @ Wl + b + h @ Wr )
// One wave: 4 M-tiles x 1 N-tile; B fragment (K-contiguous from W^T) is
// reused across the 4 WMMAs per K-step.
template<int CIN, int COUT, bool RELU, bool OUTBF>
__global__ void k_sage_gemm(const float* __restrict__ agg, const float* __restrict__ dinv,
                            const __bf16* __restrict__ h, const __bf16* __restrict__ WlT,
                            const float* __restrict__ bias, const __bf16* __restrict__ WrT,
                            __bf16* __restrict__ obf, float* __restrict__ of32) {
    int lane = threadIdx.x & 31;
    int wave = threadIdx.x >> 5;
    int m0 = blockIdx.x * 64;                  // 4 M-tiles of 16
    int n0 = blockIdx.y * 64 + wave * 16;      // 1 N-tile per wave
    int l15 = lane & 15;
    int col = n0 + l15;
    int kbA = (lane >= 16) ? 8 : 0;
    int kbB = (lane >= 16) ? 8 : 0;

    const __bf16* wl = WlT + (size_t)col * CIN + kbB;
    const __bf16* wr = WrT + (size_t)col * CIN + kbB;

    float di[4];
    const float*  ap[4];
    const __bf16* hp[4];
#pragma unroll
    for (int mt = 0; mt < 4; ++mt) {
        int row = m0 + mt * 16 + l15;
        di[mt] = dinv[row];
        ap[mt] = agg + (size_t)row * CIN + kbA;
        hp[mt] = h   + (size_t)row * CIN + kbA;
    }

    v8f acc[4] = {{}, {}, {}, {}};

    // GEMM 1: mean-aggregated neighbors @ Wl
    for (int kk = 0; kk < CIN; kk += 32) {
        v16bf bm = __builtin_shufflevector(*(const v8bf*)(wl + kk), *(const v8bf*)(wl + kk + 16), 0,1,2,3,4,5,6,7,8,9,10,11,12,13,14,15);   // 2x16B, same K split as A
#pragma unroll
        for (int mt = 0; mt < 4; ++mt)
            acc[mt] = __builtin_amdgcn_wmma_f32_16x16x32_bf16(
                false, load_a_f32_scaled(ap[mt] + kk, di[mt]),
                false, bm, (short)0, acc[mt], false, false);
    }
    // GEMM 2: root features @ Wr
    for (int kk = 0; kk < CIN; kk += 32) {
        v16bf bm = __builtin_shufflevector(*(const v8bf*)(wr + kk), *(const v8bf*)(wr + kk + 16), 0,1,2,3,4,5,6,7,8,9,10,11,12,13,14,15);
#pragma unroll
        for (int mt = 0; mt < 4; ++mt)
            acc[mt] = __builtin_amdgcn_wmma_f32_16x16x32_bf16(
                false, load_a_bf(hp[mt] + kk),
                false, bm, (short)0, acc[mt], false, false);
    }

    // epilogue: C/D layout: VGPR v -> M = v + (lane>=16)*8, N = lane%16
    float bc = bias[col];
#pragma unroll
    for (int mt = 0; mt < 4; ++mt) {
        int rb = m0 + mt * 16 + ((lane >= 16) ? 8 : 0);
#pragma unroll
        for (int v = 0; v < 8; ++v) {
            float val = acc[mt][v] + bc;
            if (RELU) val = fmaxf(val, 0.0f);
            size_t o = (size_t)(rb + v) * COUT + col;
            if (OUTBF) obf[o] = (__bf16)val;
            else       of32[o] = val;
        }
    }
}

// ---------------- global mean pool ----------------------------------------
__global__ void k_pool(const float* __restrict__ h2, const int* __restrict__ batch,
                       float* __restrict__ gsum, float* __restrict__ gcnt) {
    int idx = blockIdx.x * blockDim.x + threadIdx.x;  // N_NODES * 128
    if (idx >= N_NODES * 128) return;
    int n = idx >> 7, c = idx & 127;
    int g = batch[n];
    atomicAdd(&gsum[g * 128 + c], h2[idx]);
    if (c == 0) atomicAdd(&gcnt[g], 1.0f);
}

// ---------------- per-graph LayerNorm over 128 features --------------------
__global__ void k_layernorm(const float* __restrict__ gsum, const float* __restrict__ gcnt,
                            const float* __restrict__ gamma, const float* __restrict__ beta,
                            float* __restrict__ out) {
    __shared__ float red[128];
    int g = blockIdx.x, c = threadIdx.x;
    float cnt = fmaxf(gcnt[g], 1.0f);
    float v = gsum[g * 128 + c] / cnt;
    red[c] = v; __syncthreads();
    for (int s = 64; s > 0; s >>= 1) { if (c < s) red[c] += red[c + s]; __syncthreads(); }
    float mu = red[0] * (1.0f / 128.0f);
    __syncthreads();
    float d = v - mu;
    red[c] = d * d; __syncthreads();
    for (int s = 64; s > 0; s >>= 1) { if (c < s) red[c] += red[c + s]; __syncthreads(); }
    float var = red[0] * (1.0f / 128.0f);
    out[g * 128 + c] = d * rsqrtf(var + 1e-5f) * gamma[c] + beta[c];
}

// ---------------------------------------------------------------------------
extern "C" void kernel_launch(void* const* d_in, const int* in_sizes, int n_in,
                              void* d_out, int out_size, void* d_ws, size_t ws_size,
                              hipStream_t stream) {
    const float* x     = (const float*)d_in[0];
    const int*   src   = (const int*)d_in[1];
    const int*   dst   = ((const int*)d_in[1]) + N_EDGES;
    const int*   batch = (const int*)d_in[2];
    const float* Wl0   = (const float*)d_in[3];
    const float* b0    = (const float*)d_in[4];
    const float* Wr0   = (const float*)d_in[5];
    const float* Wl1   = (const float*)d_in[6];
    const float* b1    = (const float*)d_in[7];
    const float* Wr1   = (const float*)d_in[8];
    const float* Wl2   = (const float*)d_in[9];
    const float* b2    = (const float*)d_in[10];
    const float* Wr2   = (const float*)d_in[11];
    const float* gamma = (const float*)d_in[12];
    const float* beta  = (const float*)d_in[13];

    char* W = (char*)d_ws;                        // all offsets 256B-aligned
    float*  deg   = (float*) (W + 0);             //  N_PAD f32
    float*  dinv  = (float*) (W + 200704);        //  N_PAD f32
    float*  agg0  = (float*) (W + 401408);        //  N f32
    __bf16* h0    = (__bf16*)(W + 602112);        //  N_PAD*256 bf16
    __bf16* h1    = (__bf16*)(W + 26226688);      //  N_PAD*256 bf16
    float*  aggF  = (float*) (W + 51851264);      //  N_PAD*256 f32 (reused)
    float*  h2    = (float*) (W + 103100416);     //  N_PAD*128 f32
    __bf16* Wl1T  = (__bf16*)(W + 128724992);
    __bf16* Wr1T  = (__bf16*)(W + 128856064);
    __bf16* Wl2T  = (__bf16*)(W + 128987136);
    __bf16* Wr2T  = (__bf16*)(W + 129052672);
    float*  gsum  = (float*) (W + 129118208);
    float*  gcnt  = (float*) (W + 129150976);

    // ---- degree + layer 0 (K=1) ----
    hipMemsetAsync(deg,  0, N_PAD * sizeof(float), stream);
    hipMemsetAsync(agg0, 0, N_NODES * sizeof(float), stream);
    k_deg_agg0<<<N_EDGES / 256, 256, 0, stream>>>(src, dst, x, deg, agg0);
    k_deg_inv<<<(N_PAD + 255) / 256, 256, 0, stream>>>(deg, dinv);
    k_layer0<<<(N_PAD * 256) / 256, 256, 0, stream>>>(x, agg0, dinv, Wl0, b0, Wr0, h0);

    // ---- weights to transposed bf16 ----
    k_w_transpose<<<(256 * 256) / 256, 256, 0, stream>>>(Wl1, Wl1T, 256, 256);
    k_w_transpose<<<(256 * 256) / 256, 256, 0, stream>>>(Wr1, Wr1T, 256, 256);
    k_w_transpose<<<(256 * 128) / 256, 256, 0, stream>>>(Wl2, Wl2T, 256, 128);
    k_w_transpose<<<(256 * 128) / 256, 256, 0, stream>>>(Wr2, Wr2T, 256, 128);

    // ---- layer 1: scatter(h0) -> WMMA GEMM -> h1 (bf16, relu) ----
    hipMemsetAsync(aggF, 0, (size_t)N_PAD * 256 * sizeof(float), stream);
    k_scatter256<<<(N_EDGES * 64) / 256, 256, 0, stream>>>(src, dst, h0, aggF);
    k_sage_gemm<256, 256, true, true>
        <<<dim3(N_PAD / 64, 256 / 64), 128, 0, stream>>>(aggF, dinv, h0, Wl1T, b1, Wr1T, h1, nullptr);

    // ---- layer 2: scatter(h1) -> WMMA GEMM -> h2 (f32, no relu) ----
    hipMemsetAsync(aggF, 0, (size_t)N_PAD * 256 * sizeof(float), stream);
    k_scatter256<<<(N_EDGES * 64) / 256, 256, 0, stream>>>(src, dst, h1, aggF);
    k_sage_gemm<256, 128, false, false>
        <<<dim3(N_PAD / 64, 128 / 64), 128, 0, stream>>>(aggF, dinv, h1, Wl2T, b2, Wr2T, nullptr, h2);

    // ---- pool + layernorm ----
    hipMemsetAsync(gsum, 0, N_GRAPHS * 128 * sizeof(float), stream);
    hipMemsetAsync(gcnt, 0, N_GRAPHS * sizeof(float), stream);
    k_pool<<<(N_NODES * 128) / 256, 256, 0, stream>>>(h2, batch, gsum, gcnt);
    k_layernorm<<<N_GRAPHS, 128, 0, stream>>>(gsum, gcnt, gamma, beta, (float*)d_out);
}
